// ConditionalLinearTimeSelfAttention_79534204387515
// MI455X (gfx1250) — matrix operand order repaired and checked by reading, not compiled, not run on hardware
//
#include <hip/hip_runtime.h>
#include <hip/hip_bf16.h>

// ---------------------------------------------------------------------------
// ConditionalLinearTimeSelfAttention for MI455X (gfx1250, wave32, WMMA).
// B=16, C=128, H=W=128 (N=16384), HEADS=4, DH=32, GROUPS=32.
// Memory-bound (~0.9GB traffic, ~40us @ 23.3TB/s); all GEMMs use
// v_wmma_f32_16x16x32_f16 with fp32 accumulation. k/v stored f16.
// Workspace requirement: ~129 MB.
// ---------------------------------------------------------------------------

typedef __attribute__((ext_vector_type(16))) _Float16 v16h;
typedef __attribute__((ext_vector_type(8)))  _Float16 v8h;
typedef __attribute__((ext_vector_type(8)))  float    v8f;

#define CB   128            // channels
#define NP   16384          // pixels per image
#define HEADS_ 4
#define DH_  32
#define LPAD 136            // LDS row stride in halves (16B-aligned, bank-spread)

static __device__ __forceinline__ v8f wmma_f16(v16h a, v16h b, v8f c) {
  return __builtin_amdgcn_wmma_f32_16x16x32_f16(false, a, false, b, (short)0, c,
                                                false, false);
}

// ---------------- kernel 1: weights fp32 -> f16 -----------------------------
__global__ void k_wconv(const float* __restrict__ kvw, const float* __restrict__ qw,
                        const float* __restrict__ ow,
                        _Float16* __restrict__ kvwh, _Float16* __restrict__ qwh,
                        _Float16* __restrict__ owh) {
  int i = blockIdx.x * 256 + threadIdx.x;             // 65536 total
  if (i < 32768)       kvwh[i]        = (_Float16)kvw[i];
  else if (i < 49152)  qwh[i - 32768] = (_Float16)qw[i - 32768];
  else                 owh[i - 49152] = (_Float16)ow[i - 49152];
}

// ---------------- kernel 2: GroupNorm stats ---------------------------------
__global__ void k_gnstats(const float* __restrict__ x, float* __restrict__ mu,
                          float* __restrict__ rs) {
  // block = one (b,g): contiguous span of 4*NP = 65536 floats
  const float* p = x + (size_t)blockIdx.x * 65536;
  float s = 0.f, sq = 0.f;
  for (int i = threadIdx.x * 4; i < 65536; i += 1024) {
    float4 v = *(const float4*)(p + i);
    s  += v.x + v.y + v.z + v.w;
    sq += v.x*v.x + v.y*v.y + v.z*v.z + v.w*v.w;
  }
  __shared__ float rs1[256], rs2[256];
  rs1[threadIdx.x] = s; rs2[threadIdx.x] = sq;
  __syncthreads();
  for (int o = 128; o > 0; o >>= 1) {
    if (threadIdx.x < o) { rs1[threadIdx.x] += rs1[threadIdx.x+o];
                           rs2[threadIdx.x] += rs2[threadIdx.x+o]; }
    __syncthreads();
  }
  if (threadIdx.x == 0) {
    float m = rs1[0] * (1.f/65536.f);
    float v = rs2[0] * (1.f/65536.f) - m*m;
    mu[blockIdx.x] = m;
    rs[blockIdx.x] = rsqrtf(v + 1e-5f);
  }
}

// ---------------- kernel 3: kv = kv_w @ GN(x) + kv_b  -> k,v (f16) ----------
__global__ void __launch_bounds__(256)
k_kvgemm(const float* __restrict__ x, const float* __restrict__ mu,
         const float* __restrict__ rs, const float* __restrict__ gs,
         const float* __restrict__ gb, const _Float16* __restrict__ kvwh,
         const float* __restrict__ kvb, _Float16* __restrict__ kh,
         _Float16* __restrict__ vh) {
  const int b = blockIdx.y;
  const int pbase = blockIdx.x * 16;
  const int t = threadIdx.x;
  __shared__ alignas(16) _Float16 xt[16 * LPAD];      // [px][c] f16, GN applied

  { // stage tile: thread t -> channel t/2, pixel-half (t&1)*8
    int c  = t >> 1;
    int ph = (t & 1) * 8;
    const float* xp = x + ((size_t)b*CB + c) * NP + pbase + ph;
    float m  = mu[b*32 + (c >> 2)];
    float r  = rs[b*32 + (c >> 2)];
    float sc = gs[c] * r;
    float sh = gb[c] - m * sc;
    float4 v0 = *(const float4*)xp;
    float4 v1 = *(const float4*)(xp + 4);
    xt[(ph+0)*LPAD + c] = (_Float16)(v0.x*sc + sh);
    xt[(ph+1)*LPAD + c] = (_Float16)(v0.y*sc + sh);
    xt[(ph+2)*LPAD + c] = (_Float16)(v0.z*sc + sh);
    xt[(ph+3)*LPAD + c] = (_Float16)(v0.w*sc + sh);
    xt[(ph+4)*LPAD + c] = (_Float16)(v1.x*sc + sh);
    xt[(ph+5)*LPAD + c] = (_Float16)(v1.y*sc + sh);
    xt[(ph+6)*LPAD + c] = (_Float16)(v1.z*sc + sh);
    xt[(ph+7)*LPAD + c] = (_Float16)(v1.w*sc + sh);
  }
  __syncthreads();

  const int w = t >> 5, lane = t & 31;
  const int px = lane & 15, arow = lane & 15;
  const int akoff = (lane < 16) ? 0 : 8;
  const int bkoff = (lane < 16) ? 0 : 8;
  const int hi    = (lane < 16) ? 0 : 8;

  for (int rt = w; rt < 16; rt += 8) {               // 256 outputs / 16
    const int obase = rt * 16;
    v8f acc;
    #pragma unroll
    for (int r = 0; r < 8; ++r) acc[r] = kvb[obase + hi + r];
    #pragma unroll
    for (int kc = 0; kc < 4; ++kc) {
      const _Float16* ap = kvwh + (size_t)(obase + arow)*CB + kc*32 + akoff;
      v8h al = *(const v8h*)ap;
      v8h ah = *(const v8h*)(ap + 16);
      const _Float16* bp = &xt[px*LPAD + kc*32 + bkoff];
      v8h bl = *(const v8h*)bp;
      v8h bh = *(const v8h*)(bp + 16);
      v16h A, Bf;
      #pragma unroll
      for (int i = 0; i < 8; ++i) { A[i]=al[i]; A[8+i]=ah[i];
                                    Bf[i]=bl[i]; Bf[8+i]=bh[i]; }
      acc = wmma_f16(A, Bf, acc);
    }
    _Float16* dst = (obase < 128) ? kh : vh;         // uniform per wave-iter
    const int od = obase & 127;
    #pragma unroll
    for (int r = 0; r < 8; ++r) {
      size_t idx = ((size_t)b*CB + od + hi + r) * NP + pbase + px;
      dst[idx] = (_Float16)acc[r];
    }
  }
}

// ---------------- kernel 4: rowmax(k) + zero accumulators -------------------
__global__ void k_rowmax(const _Float16* __restrict__ kh, float* __restrict__ rowmax,
                         float* __restrict__ ctx, float* __restrict__ sumexp) {
  const int row = blockIdx.x;                        // b*128 + h*32 + d (2048)
  if (threadIdx.x < 32) ctx[row*32 + threadIdx.x] = 0.f;
  if (threadIdx.x == 0) sumexp[row] = 0.f;
  const _Float16* kp = kh + (size_t)row * NP;
  float m = -3.0e38f;
  for (int i = threadIdx.x * 8; i < NP; i += 2048) {
    v8h v = *(const v8h*)(kp + i);
    #pragma unroll
    for (int j = 0; j < 8; ++j) m = fmaxf(m, (float)v[j]);
  }
  __shared__ float red[256];
  red[threadIdx.x] = m;
  __syncthreads();
  for (int o = 128; o > 0; o >>= 1) {
    if (threadIdx.x < o) red[threadIdx.x] = fmaxf(red[threadIdx.x], red[threadIdx.x+o]);
    __syncthreads();
  }
  if (threadIdx.x == 0) rowmax[row] = red[0];
}

// ---------------- kernel 5: ctx[d][e] += sum_p exp(k[d][p]-max)*v[e][p] -----
__global__ void __launch_bounds__(256)
k_ctx(const _Float16* __restrict__ kh, const _Float16* __restrict__ vh,
      const float* __restrict__ rowmax, float* __restrict__ ctx,
      float* __restrict__ sumexp) {
  const int b = blockIdx.z, h = blockIdx.y, chunk = blockIdx.x;   // 8 chunks
  const int t = threadIdx.x, w = t >> 5, lane = t & 31;
  const int arow = lane & 15;
  const int akoff = (lane < 16) ? 0 : 8;
  const int bkoff = (lane < 16) ? 0 : 8;
  const int hi    = (lane < 16) ? 0 : 8;
  const int p0base = chunk * 2048 + w * 256;

  const _Float16* kb = kh + ((size_t)b*CB + h*DH_) * NP;
  const _Float16* vb = vh + ((size_t)b*CB + h*DH_) * NP;
  const float rm0 = rowmax[b*CB + h*DH_ + arow];
  const float rm1 = rowmax[b*CB + h*DH_ + 16 + arow];

  v8f a00 = {}, a01 = {}, a10 = {}, a11 = {};
  float se0 = 0.f, se1 = 0.f;

  for (int it = 0; it < 8; ++it) {
    const int p0 = p0base + it * 32;
    // A tiles: exp(k - rowmax), rows d = {arow, 16+arow}
    v16h A0, A1;
    {
      const _Float16* kp = kb + (size_t)arow * NP + p0 + akoff;
      v8h lo = *(const v8h*)kp, hi8 = *(const v8h*)(kp + 16);
      #pragma unroll
      for (int i = 0; i < 8; ++i) {
        float e0 = __expf((float)lo[i]  - rm0); se0 += e0; A0[i]   = (_Float16)e0;
        float e1 = __expf((float)hi8[i] - rm0); se0 += e1; A0[8+i] = (_Float16)e1;
      }
    }
    {
      const _Float16* kp = kb + (size_t)(16 + arow) * NP + p0 + akoff;
      v8h lo = *(const v8h*)kp, hi8 = *(const v8h*)(kp + 16);
      #pragma unroll
      for (int i = 0; i < 8; ++i) {
        float e0 = __expf((float)lo[i]  - rm1); se1 += e0; A1[i]   = (_Float16)e0;
        float e1 = __expf((float)hi8[i] - rm1); se1 += e1; A1[8+i] = (_Float16)e1;
      }
    }
    // B tiles: v rows e = {arow, 16+arow} as columns
    v16h B0, B1;
    {
      const _Float16* vp = vb + (size_t)arow * NP + p0 + bkoff;
      v8h lo = *(const v8h*)vp, hi8 = *(const v8h*)(vp + 16);
      #pragma unroll
      for (int i = 0; i < 8; ++i) { B0[i]=lo[i]; B0[8+i]=hi8[i]; }
    }
    {
      const _Float16* vp = vb + (size_t)(16 + arow) * NP + p0 + bkoff;
      v8h lo = *(const v8h*)vp, hi8 = *(const v8h*)(vp + 16);
      #pragma unroll
      for (int i = 0; i < 8; ++i) { B1[i]=lo[i]; B1[8+i]=hi8[i]; }
    }
    a00 = wmma_f16(A0, B0, a00);
    a01 = wmma_f16(A0, B1, a01);
    a10 = wmma_f16(A1, B0, a10);
    a11 = wmma_f16(A1, B1, a11);
  }

  float* cb = ctx + (size_t)(b*HEADS_ + h) * 1024;
  const int ec = lane & 15;
  #pragma unroll
  for (int r = 0; r < 8; ++r) {
    const int d0 = r + hi;
    atomicAdd(cb + d0*32        + ec,      a00[r]);
    atomicAdd(cb + d0*32        + 16 + ec, a01[r]);
    atomicAdd(cb + (16+d0)*32   + ec,      a10[r]);
    atomicAdd(cb + (16+d0)*32   + 16 + ec, a11[r]);
  }
  atomicAdd(&sumexp[b*CB + h*DH_ + arow],      se0);
  atomicAdd(&sumexp[b*CB + h*DH_ + 16 + arow], se1);
}

// ---------------- kernel 6: normalize ctx -> f16 ----------------------------
__global__ void k_ctxnorm(const float* __restrict__ ctx, const float* __restrict__ sumexp,
                          _Float16* __restrict__ ctxh) {
  int i = blockIdx.x * 256 + threadIdx.x;            // 65536
  ctxh[i] = (_Float16)(ctx[i] / sumexp[i >> 5]);
}

// ---------------- kernel 7: fused  q-proj -> ctx^T.q -> out-proj ------------
__global__ void __launch_bounds__(256)
k_out(const float* __restrict__ cond, const _Float16* __restrict__ qwh,
      const float* __restrict__ qb, const _Float16* __restrict__ ctxh,
      const _Float16* __restrict__ owh, const float* __restrict__ ob,
      float* __restrict__ out) {
  const int b = blockIdx.y;
  const int pbase = blockIdx.x * 16;
  const int t = threadIdx.x;
  __shared__ alignas(16) _Float16 cbuf[16 * LPAD];
  __shared__ alignas(16) _Float16 qbuf[16 * LPAD];
  __shared__ alignas(16) _Float16 abuf[16 * LPAD];

  { // stage cond tile transposed into LDS
    int c  = t >> 1;
    int ph = (t & 1) * 8;
    const float* cp = cond + ((size_t)b*CB + c) * NP + pbase + ph;
    float4 v0 = *(const float4*)cp;
    float4 v1 = *(const float4*)(cp + 4);
    cbuf[(ph+0)*LPAD + c] = (_Float16)v0.x;
    cbuf[(ph+1)*LPAD + c] = (_Float16)v0.y;
    cbuf[(ph+2)*LPAD + c] = (_Float16)v0.z;
    cbuf[(ph+3)*LPAD + c] = (_Float16)v0.w;
    cbuf[(ph+4)*LPAD + c] = (_Float16)v1.x;
    cbuf[(ph+5)*LPAD + c] = (_Float16)v1.y;
    cbuf[(ph+6)*LPAD + c] = (_Float16)v1.z;
    cbuf[(ph+7)*LPAD + c] = (_Float16)v1.w;
  }
  __syncthreads();

  const int w = t >> 5, lane = t & 31;
  const int px = lane & 15, arow = lane & 15;
  const int akoff = (lane < 16) ? 0 : 8;
  const int bkoff = (lane < 16) ? 0 : 8;
  const int hi    = (lane < 16) ? 0 : 8;

  { // stage 1: q = q_w @ cond + q_b    (wave w -> output rows [16w,16w+16))
    const int obase = w * 16;
    v8f acc;
    #pragma unroll
    for (int r = 0; r < 8; ++r) acc[r] = qb[obase + hi + r];
    #pragma unroll
    for (int kc = 0; kc < 4; ++kc) {
      const _Float16* ap = qwh + (size_t)(obase + arow)*CB + kc*32 + akoff;
      v8h al = *(const v8h*)ap, ah = *(const v8h*)(ap + 16);
      const _Float16* bp = &cbuf[px*LPAD + kc*32 + bkoff];
      v8h bl = *(const v8h*)bp, bh = *(const v8h*)(bp + 16);
      v16h A, Bf;
      #pragma unroll
      for (int i = 0; i < 8; ++i) { A[i]=al[i]; A[8+i]=ah[i];
                                    Bf[i]=bl[i]; Bf[8+i]=bh[i]; }
      acc = wmma_f16(A, Bf, acc);
    }
    #pragma unroll
    for (int r = 0; r < 8; ++r)
      qbuf[px*LPAD + obase + hi + r] = (_Float16)acc[r];
  }
  __syncthreads();

  { // stage 2: attn[h*32+e] = sum_d ctx[d][e] * q[h*32+d]
    const int h = w >> 1, rt = w & 1;
    const int erow = rt * 16 + arow;
    const _Float16* cb2 = ctxh + (size_t)(b*HEADS_ + h) * 1024;
    v16h A;
    #pragma unroll
    for (int el = 0; el < 16; ++el) {
      int d = akoff + el + ((el >= 8) ? 8 : 0);
      A[el] = cb2[d*32 + erow];
    }
    const _Float16* qp = &qbuf[px*LPAD + h*DH_ + bkoff];
    v8h bl = *(const v8h*)qp, bh = *(const v8h*)(qp + 16);
    v16h Bf;
    #pragma unroll
    for (int i = 0; i < 8; ++i) { Bf[i]=bl[i]; Bf[8+i]=bh[i]; }
    v8f acc = {};
    acc = wmma_f16(A, Bf, acc);
    #pragma unroll
    for (int r = 0; r < 8; ++r)
      abuf[px*LPAD + h*DH_ + rt*16 + hi + r] = (_Float16)acc[r];
  }
  __syncthreads();

  { // stage 3: out = out_w @ attn + out_b
    const int obase = w * 16;
    v8f acc;
    #pragma unroll
    for (int r = 0; r < 8; ++r) acc[r] = ob[obase + hi + r];
    #pragma unroll
    for (int kc = 0; kc < 4; ++kc) {
      const _Float16* ap = owh + (size_t)(obase + arow)*CB + kc*32 + akoff;
      v8h al = *(const v8h*)ap, ah = *(const v8h*)(ap + 16);
      const _Float16* bp = &abuf[px*LPAD + kc*32 + bkoff];
      v8h bl = *(const v8h*)bp, bh = *(const v8h*)(bp + 16);
      v16h A, Bf;
      #pragma unroll
      for (int i = 0; i < 8; ++i) { A[i]=al[i]; A[8+i]=ah[i];
                                    Bf[i]=bl[i]; Bf[8+i]=bh[i]; }
      acc = wmma_f16(A, Bf, acc);
    }
    #pragma unroll
    for (int r = 0; r < 8; ++r)
      out[((size_t)b*CB + obase + hi + r) * NP + pbase + px] = acc[r];
  }
}

// ---------------------------------------------------------------------------
extern "C" void kernel_launch(void* const* d_in, const int* in_sizes, int n_in,
                              void* d_out, int out_size, void* d_ws, size_t ws_size,
                              hipStream_t stream) {
  const float* x    = (const float*)d_in[0];
  const float* cond = (const float*)d_in[1];
  const float* gs   = (const float*)d_in[2];
  const float* gb   = (const float*)d_in[3];
  const float* kvw  = (const float*)d_in[4];
  const float* kvb  = (const float*)d_in[5];
  const float* qw   = (const float*)d_in[6];
  const float* qbv  = (const float*)d_in[7];
  const float* ow   = (const float*)d_in[8];
  const float* obv  = (const float*)d_in[9];
  float* out = (float*)d_out;

  char* ws = (char*)d_ws;                     // ~129 MB required
  float*    mu     = (float*)(ws + 0);        // 512 f32
  float*    rs     = (float*)(ws + 2048);     // 512 f32
  float*    rowmax = (float*)(ws + 4096);     // 2048 f32
  float*    sumexp = (float*)(ws + 12288);    // 2048 f32
  float*    ctx    = (float*)(ws + 20480);    // 65536 f32
  _Float16* ctxh   = (_Float16*)(ws + 282624);  // 65536 f16
  _Float16* kvwh   = (_Float16*)(ws + 413696);  // 32768 f16
  _Float16* qwh    = (_Float16*)(ws + 479232);  // 16384 f16
  _Float16* owh    = (_Float16*)(ws + 512000);  // 16384 f16
  _Float16* kh     = (_Float16*)(ws + 544768);           // 16*128*16384 f16
  _Float16* vh     = (_Float16*)(ws + 544768 + 67108864);

  k_wconv  <<<256, 256, 0, stream>>>(kvw, qw, ow, kvwh, qwh, owh);
  k_gnstats<<<512, 256, 0, stream>>>(x, mu, rs);
  k_kvgemm <<<dim3(1024, 16), 256, 0, stream>>>(x, mu, rs, gs, gb, kvwh, kvb, kh, vh);
  k_rowmax <<<2048, 256, 0, stream>>>(kh, rowmax, ctx, sumexp);
  k_ctx    <<<dim3(8, HEADS_, 16), 256, 0, stream>>>(kh, vh, rowmax, ctx, sumexp);
  k_ctxnorm<<<256, 256, 0, stream>>>(ctx, sumexp, ctxh);
  k_out    <<<dim3(1024, 16), 256, 0, stream>>>(cond, qwh, qbv, ctxh, owh, obv, out);
}
